// KmeanModel_35553739276688
// MI455X (gfx1250) — matrix operand order repaired and checked by reading, not compiled, not run on hardware
//
#include <hip/hip_runtime.h>

// WMMA operand vector types (probe-verified builtin signature)
typedef __attribute__((ext_vector_type(16))) __bf16 v16bf;
typedef __attribute__((ext_vector_type(8)))  float  v8f;

namespace {
constexpr int kN = 8192;   // points
constexpr int kK = 128;    // centroids
constexpr int kD = 512;    // dim
constexpr int kMRows = 32; // rows (points) per workgroup
// LDS row stride (bf16 elements): 512 + 8 pad -> row-to-row advance of 260
// dwords = 4 banks, so the 16 lanes of a ds_load_b128 column read hit all 64
// banks conflict-free. 520*2 bytes keeps 16B alignment.
constexpr int kLdsStride = 520;
}

struct alignas(16) U128 { unsigned int w0, w1, w2, w3; };
struct U128x2 { U128 a, b; };   // 32 bytes == v16bf

static __device__ inline v16bf load_frag(const __bf16* base0, const __bf16* base1) {
    U128x2 p;
    p.a = *(const U128*)base0;   // b128: 8 bf16
    p.b = *(const U128*)base1;   // b128: 8 bf16
    return __builtin_bit_cast(v16bf, p);
}

// ---------------------------------------------------------------------------
// Kernel 1: centroids f32 [128,512] -> bf16 hi plane + bf16 lo plane + norms.
// Runs once per launch; 256 KB output stays hot in L2 for the main kernel.
// ---------------------------------------------------------------------------
__global__ __launch_bounds__(256)
void centroid_prep(const float* __restrict__ cent,
                   __bf16* __restrict__ ch, __bf16* __restrict__ cl,
                   float* __restrict__ cnorm)
{
    __shared__ float norm[16];
    const int tid = threadIdx.x;
    if (tid < 16) norm[tid] = 0.0f;
    __syncthreads();

    const int r  = blockIdx.x * 16 + (tid >> 4);   // centroid row
    const int c0 = (tid & 15) * 32;                // 32 contiguous f32
    const float* src = cent + (size_t)r * kD + c0;
    __bf16* dh = ch + (size_t)r * kD + c0;
    __bf16* dl = cl + (size_t)r * kD + c0;

    float s2 = 0.0f;
#pragma unroll
    for (int j = 0; j < 32; j += 4) {
        float4 v4 = *(const float4*)(src + j);
        float f[4] = {v4.x, v4.y, v4.z, v4.w};
#pragma unroll
        for (int e = 0; e < 4; ++e) {
            __bf16 hi = (__bf16)f[e];                 // RNE
            dh[j + e] = hi;
            dl[j + e] = (__bf16)(f[e] - (float)hi);   // residual
            s2 += f[e] * f[e];
        }
    }
    atomicAdd(&norm[tid >> 4], s2);
    __syncthreads();
    if (tid < 16) cnorm[blockIdx.x * 16 + tid] = norm[tid];
}

// ---------------------------------------------------------------------------
// Kernel 2: dist[n,k] = sqrt(||x_n||^2 + ||c_k||^2 - 2 x_n.c_k) via bf16
// hi/lo-split WMMA (hi*hi + lo*hi + hi*lo) with f32 accumulation.
// 32-row M-tile per workgroup: B fragments reused across two row strips,
// two independent accumulator chains per wave.
// ---------------------------------------------------------------------------
__global__ __launch_bounds__(256)
void kmeans_dist_wmma(const float* __restrict__ x,
                      const __bf16* __restrict__ ch,
                      const __bf16* __restrict__ cl,
                      const float* __restrict__ cnorm,
                      float* __restrict__ out)
{
    __shared__ __bf16 Ah[kMRows][kLdsStride];   // x strip, bf16 high part
    __shared__ __bf16 Al[kMRows][kLdsStride];   // x strip, bf16 residual part
    __shared__ float  xnorm[kMRows];            // per-row ||x||^2 (full f32)

    const int tid  = threadIdx.x;
    const int lane = tid & 31;
    const int wave = tid >> 5;        // 8 waves: one 16-centroid tile each
    const int half = lane >> 4;       // 0 | 1 (wave half, per ISA A/B layouts)
    const int l16  = lane & 15;
    const int row_base = blockIdx.x * kMRows;

    if (tid < kMRows) xnorm[tid] = 0.0f;
    __syncthreads();

    // ---- Stage 32x512 f32 x-strip into LDS as bf16 hi + lo; row norms in f32.
    {
        const int r  = tid >> 3;          // row 0..31 (8 threads per row)
        const int c0 = (tid & 7) * 64;    // 64 contiguous f32 per thread
        const float* src = x + (size_t)(row_base + r) * kD + c0;
        float sum2 = 0.0f;
#pragma unroll
        for (int j = 0; j < 64; j += 4) {
            float4 v4 = *(const float4*)(src + j);
            float f[4] = {v4.x, v4.y, v4.z, v4.w};
#pragma unroll
            for (int e = 0; e < 4; ++e) {
                __bf16 hi = (__bf16)f[e];
                Ah[r][c0 + j + e] = hi;
                Al[r][c0 + j + e] = (__bf16)(f[e] - (float)hi);
                sum2 += f[e] * f[e];
            }
        }
        atomicAdd(&xnorm[r], sum2);       // ds_add_f32
    }
    __syncthreads();

    // ---- Per-wave GEMM tile: 32 rows x 16 centroids, K-loop over D.
    const int crow = wave * 16 + l16;     // this lane's centroid (B column)
    // B fragment: lanes 0-15 hold K 0..7 & 16..23; lanes 16-31 hold 8..15 & 24..31.
    const __bf16* bh = ch + (size_t)crow * kD + half * 8;
    const __bf16* bl = cl + (size_t)crow * kD + half * 8;
    const float cnrm = cnorm[crow];

    // A fragment: lanes 0-15 hold K 0..7 & 16..23; lanes 16-31 hold 8..15 & 24..31.
    const int offA = half * 8;
    const int r0 = l16;          // strip 0 row
    const int r1 = l16 + 16;     // strip 1 row

    v8f acc0 = {};               // rows 0..15  (f32 C/D 16x16)
    v8f acc1 = {};               // rows 16..31

#pragma unroll 2
    for (int kb = 0; kb < kD; kb += 32) {
        v16bf a0h = load_frag(&Ah[r0][kb + offA], &Ah[r0][kb + offA + 16]);
        v16bf a0l = load_frag(&Al[r0][kb + offA], &Al[r0][kb + offA + 16]);
        v16bf a1h = load_frag(&Ah[r1][kb + offA], &Ah[r1][kb + offA + 16]);
        v16bf a1l = load_frag(&Al[r1][kb + offA], &Al[r1][kb + offA + 16]);
        v16bf bvh = load_frag(bh + kb, bh + kb + 16);
        v16bf bvl = load_frag(bl + kb, bl + kb + 16);

        // x . c via hi/lo split: hi*hi + lo*hi + hi*lo (lo*lo negligible)
        acc0 = __builtin_amdgcn_wmma_f32_16x16x32_bf16(false, a0h, false, bvh,
                                                       (short)0, acc0, false, false);
        acc1 = __builtin_amdgcn_wmma_f32_16x16x32_bf16(false, a1h, false, bvh,
                                                       (short)0, acc1, false, false);
        acc0 = __builtin_amdgcn_wmma_f32_16x16x32_bf16(false, a0l, false, bvh,
                                                       (short)0, acc0, false, false);
        acc1 = __builtin_amdgcn_wmma_f32_16x16x32_bf16(false, a1l, false, bvh,
                                                       (short)0, acc1, false, false);
        acc0 = __builtin_amdgcn_wmma_f32_16x16x32_bf16(false, a0h, false, bvl,
                                                       (short)0, acc0, false, false);
        acc1 = __builtin_amdgcn_wmma_f32_16x16x32_bf16(false, a1h, false, bvl,
                                                       (short)0, acc1, false, false);
    }

    // C/D layout: VGPR i, lanes 0-15 -> M=i, N=lane; lanes 16-31 -> M=i+8.
    const int m0 = half * 8;
#pragma unroll
    for (int i = 0; i < 8; ++i) {
        float d2a = xnorm[m0 + i] + cnrm - 2.0f * acc0[i];
        d2a = d2a < 0.0f ? 0.0f : d2a;
        out[(size_t)(row_base + m0 + i) * kK + crow] = __builtin_sqrtf(d2a);

        float d2b = xnorm[16 + m0 + i] + cnrm - 2.0f * acc1[i];
        d2b = d2b < 0.0f ? 0.0f : d2b;
        out[(size_t)(row_base + 16 + m0 + i) * kK + crow] = __builtin_sqrtf(d2b);
    }
}

extern "C" void kernel_launch(void* const* d_in, const int* in_sizes, int n_in,
                              void* d_out, int out_size, void* d_ws, size_t ws_size,
                              hipStream_t stream) {
    (void)in_sizes; (void)n_in; (void)out_size; (void)ws_size;
    const float* x    = (const float*)d_in[0];
    const float* cent = (const float*)d_in[1];
    float* out        = (float*)d_out;

    // Workspace layout: bf16 hi plane | bf16 lo plane | f32 centroid norms
    char* ws = (char*)d_ws;
    __bf16* ch   = (__bf16*)ws;                                    // 128 KB
    __bf16* cl   = (__bf16*)(ws + (size_t)kK * kD * sizeof(__bf16));
    float*  cnrm = (float*)(ws + 2 * (size_t)kK * kD * sizeof(__bf16));

    hipLaunchKernelGGL(centroid_prep, dim3(kK / 16), dim3(256), 0, stream,
                       cent, ch, cl, cnrm);
    hipLaunchKernelGGL(kmeans_dist_wmma, dim3(kN / kMRows), dim3(256), 0, stream,
                       x, ch, cl, cnrm, out);
}
